// StyleTransformer_75986561401014
// MI455X (gfx1250) — matrix operand order repaired and checked by reading, not compiled, not run on hardware
//
#include <hip/hip_runtime.h>
#include <hip/hip_bf16.h>

// ---------------------------------------------------------------------------
// StyleTransformer fused MoE-MLP for gfx1250 (MI455X), f16 WMMA, f32 accum.
// ---------------------------------------------------------------------------

#define D 512
#define BT 64            // rows per tile
#define LSTR 520         // LDS row stride in halves (pad to break bank conflicts)
#define BATCH 16384
#define NEXP 8
#define MAXTILES 256     // worst case: all rows in one expert

typedef _Float16 h8_t   __attribute__((ext_vector_type(8)));
typedef _Float16 h16_t  __attribute__((ext_vector_type(16)));
typedef float    f32x8_t __attribute__((ext_vector_type(8)));

// ---------------------------------------------------------------------------
// Core GEMM: out[BT x D] (+)= in[BT x D] * W^T  using v_wmma_f32_16x16x32_f16
// Block = 256 threads = 8 waves. Wave w owns output columns [w*64, w*64+64).
// acc[rt][ct] are 16x16 f32 tiles (8 VGPRs each).
// ---------------------------------------------------------------------------
__device__ __forceinline__ void gemm_core(const _Float16* __restrict__ in_lds,
                                          const _Float16* __restrict__ W,
                                          f32x8_t acc[4][4]) {
    const int tid  = threadIdx.x;
    const int wv   = tid >> 5;
    const int lane = tid & 31;
    const int lrow = lane & 15;
    const int hi   = lane >> 4;          // 0 or 1 (lane half)
    const int obase = wv * 64;

    for (int kk = 0; kk < D; kk += 32) {
        // B fragments: lane n holds column (obase+ct*16+lrow),
        // K in {kk+hi*8 .. +7} and {kk+16+hi*8 .. +7}
        h16_t bfr[4];
#pragma unroll
        for (int ct = 0; ct < 4; ++ct) {
            const _Float16* p = W + (size_t)(obase + ct * 16 + lrow) * D + kk + hi * 8;
            h8_t b0 = *(const h8_t*)(p);
            h8_t b1 = *(const h8_t*)(p + 16);
            bfr[ct] = __builtin_shufflevector(b0, b1, 0,1,2,3,4,5,6,7,8,9,10,11,12,13,14,15);
        }
        // A fragments: lane l holds row (rt*16+lrow), K in {kk+hi*8 .. +7} and {kk+16+hi*8 .. +7}
#pragma unroll
        for (int rt = 0; rt < 4; ++rt) {
            const _Float16* p = in_lds + (rt * 16 + lrow) * LSTR + kk + hi * 8;
            h8_t a0 = *(const h8_t*)(p);
            h8_t a1 = *(const h8_t*)(p + 16);
            h16_t afr = __builtin_shufflevector(a0, a1, 0,1,2,3,4,5,6,7,8,9,10,11,12,13,14,15);
#pragma unroll
            for (int ct = 0; ct < 4; ++ct) {
                acc[rt][ct] = __builtin_amdgcn_wmma_f32_16x16x32_f16(
                    false, afr, false, bfr[ct], (short)0, acc[rt][ct], false, false);
            }
        }
    }
}

// Epilogue: bias (+ optional ReLU), convert to f16, store to next LDS buffer.
__device__ __forceinline__ void mlp_layer(const _Float16* __restrict__ in_lds,
                                          _Float16* __restrict__ out_lds,
                                          const _Float16* __restrict__ W,
                                          const float* __restrict__ bias_lds,
                                          bool relu) {
    const int lane = threadIdx.x & 31;
    const int wv   = threadIdx.x >> 5;
    const int lrow = lane & 15;
    const int hi   = lane >> 4;
    const int obase = wv * 64;

    f32x8_t acc[4][4] = {};
    gemm_core(in_lds, W, acc);

#pragma unroll
    for (int rt = 0; rt < 4; ++rt) {
#pragma unroll
        for (int ct = 0; ct < 4; ++ct) {
            const int col   = obase + ct * 16 + lrow;
            const int rbase = rt * 16 + hi * 8;
            const float bv  = bias_lds[col];
#pragma unroll
            for (int v = 0; v < 8; ++v) {
                float x = acc[rt][ct][v] + bv;
                if (relu) x = fmaxf(x, 0.0f);
                out_lds[(rbase + v) * LSTR + col] = (_Float16)x;
            }
        }
    }
}

// ---------------------------------------------------------------------------
// Kernel 1: f32 -> f16 weight conversion
// ---------------------------------------------------------------------------
__global__ void st_cvt_f16(const float* __restrict__ in, _Float16* __restrict__ out, int n) {
    int i = blockIdx.x * 256 + threadIdx.x;
    if (i < n) out[i] = (_Float16)in[i];
}

// ---------------------------------------------------------------------------
// Kernel 2: partition rows by expert id (single block, LDS histogram+scan)
// off[0..8] = exclusive prefix; idxlist[off[e]..off[e+1]) = rows of expert e.
// ---------------------------------------------------------------------------
__global__ __launch_bounds__(1024)
void st_partition(const int* __restrict__ y, int* __restrict__ off, int* __restrict__ idxlist) {
    __shared__ int cnt[NEXP];
    const int tid = threadIdx.x;
    if (tid < NEXP) cnt[tid] = 0;
    __syncthreads();
    for (int i = tid; i < BATCH; i += 1024) atomicAdd(&cnt[y[i]], 1);
    __syncthreads();
    if (tid == 0) {
        int run = 0;
        for (int e = 0; e < NEXP; ++e) {
            int c = cnt[e];
            off[e] = run;
            cnt[e] = run;     // reuse as scatter cursor
            run += c;
        }
        off[NEXP] = run;
    }
    __syncthreads();
    for (int i = tid; i < BATCH; i += 1024) {
        int pos = atomicAdd(&cnt[y[i]], 1);
        idxlist[pos] = i;
    }
}

// ---------------------------------------------------------------------------
// Kernel 3: fused 4-layer decomposer for one 64-row tile.
// Writes s_unrelated (f32) and s_related (f16) to workspace.
// ---------------------------------------------------------------------------
__global__ __launch_bounds__(256)
void st_decomposer(const float* __restrict__ s,
                   const _Float16* __restrict__ W16,
                   const float* __restrict__ dec_b,
                   float* __restrict__ sunrel,
                   _Float16* __restrict__ srel) {
    __shared__ _Float16 buf[2][BT * LSTR];
    __shared__ float biasL[4 * D];
    const int tid  = threadIdx.x;
    const int row0 = blockIdx.x * BT;

    for (int i = tid; i < 4 * D; i += 256) biasL[i] = dec_b[i];
    for (int i = tid; i < BT * D; i += 256) {
        int r = i >> 9, c = i & (D - 1);
        buf[0][r * LSTR + c] = (_Float16)s[(size_t)(row0 + r) * D + c];
    }
    __syncthreads();

    int cur = 0;
    for (int l = 0; l < 4; ++l) {
        mlp_layer(buf[cur], buf[cur ^ 1], W16 + (size_t)l * D * D, biasL + l * D, true);
        cur ^= 1;
        __syncthreads();
    }
    // h4 (= s_unrelated) is in buf[cur] (cur == 0 after 4 flips)
    for (int i = tid; i < BT * D; i += 256) {
        int r = i >> 9, c = i & (D - 1);
        float h = (float)buf[cur][r * LSTR + c];
        size_t g = (size_t)(row0 + r) * D + c;
        sunrel[g] = h;
        srel[g]   = (_Float16)(s[g] - h);
    }
}

// ---------------------------------------------------------------------------
// Kernel 4: fused 4-layer expert MLP on gathered rows of its expert.
// Final layer adds bias + s_unrelated and scatters directly into d_out.
// ---------------------------------------------------------------------------
__global__ __launch_bounds__(256)
void st_expert(const _Float16* __restrict__ srel,
               const float* __restrict__ sunrel,
               const _Float16* __restrict__ W16exp,
               const float* __restrict__ exp_b,
               const int* __restrict__ off,
               const int* __restrict__ idxlist,
               float* __restrict__ out) {
    const int e = blockIdx.x >> 8;        // MAXTILES == 256
    const int t = blockIdx.x & (MAXTILES - 1);
    const int beg = off[e], end = off[e + 1];
    const int start = beg + t * BT;
    if (start >= end) return;             // workgroup-uniform exit
    const int nrows = min(BT, end - start);

    __shared__ _Float16 buf[2][BT * LSTR];
    __shared__ float biasL[4 * D];
    __shared__ int rowidx[BT];
    const int tid = threadIdx.x;

    for (int i = tid; i < 4 * D; i += 256) biasL[i] = exp_b[(size_t)e * 4 * D + i];
    for (int i = tid; i < BT; i += 256)    rowidx[i] = (i < nrows) ? idxlist[start + i] : 0;
    __syncthreads();
    for (int i = tid; i < BT * D; i += 256) {
        int r = i >> 9, c = i & (D - 1);
        buf[0][r * LSTR + c] = (r < nrows) ? srel[(size_t)rowidx[r] * D + c] : (_Float16)0.0f;
    }
    __syncthreads();

    const _Float16* We = W16exp + (size_t)e * 4 * D * D;
    int cur = 0;
    for (int l = 0; l < 3; ++l) {
        mlp_layer(buf[cur], buf[cur ^ 1], We + (size_t)l * D * D, biasL + l * D, true);
        cur ^= 1;
        __syncthreads();
    }
    // Final layer: no ReLU; add s_unrelated and scatter to output rows.
    {
        f32x8_t acc[4][4] = {};
        gemm_core(buf[cur], We + (size_t)3 * D * D, acc);

        const int lane = tid & 31;
        const int wv   = tid >> 5;
        const int lrow = lane & 15;
        const int hi   = lane >> 4;
        const int obase = wv * 64;
        const float* bias3 = biasL + 3 * D;
#pragma unroll
        for (int rt = 0; rt < 4; ++rt) {
#pragma unroll
            for (int ct = 0; ct < 4; ++ct) {
                const int col   = obase + ct * 16 + lrow;
                const int rbase = rt * 16 + hi * 8;
                const float bv  = bias3[col];
#pragma unroll
                for (int v = 0; v < 8; ++v) {
                    const int rl = rbase + v;
                    if (rl < nrows) {
                        size_t g = (size_t)rowidx[rl] * D + col;
                        out[g] = acc[rt][ct][v] + bv + sunrel[g];
                    }
                }
            }
        }
    }
}

// ---------------------------------------------------------------------------
// Launch
// ---------------------------------------------------------------------------
extern "C" void kernel_launch(void* const* d_in, const int* in_sizes, int n_in,
                              void* d_out, int out_size, void* d_ws, size_t ws_size,
                              hipStream_t stream) {
    const float* s    = (const float*)d_in[0];
    const int*   y    = (const int*)d_in[1];
    const float* decW = (const float*)d_in[2];
    const float* decb = (const float*)d_in[3];
    const float* expW = (const float*)d_in[4];
    const float* expb = (const float*)d_in[5];
    float* out = (float*)d_out;

    char* ws = (char*)d_ws;
    const size_t MB = 1024ull * 1024ull;
    _Float16* W16dec = (_Float16*)(ws + 0);          //  2 MiB  (4*512*512 f16)
    _Float16* W16exp = (_Float16*)(ws + 2  * MB);    // 16 MiB  (8*4*512*512 f16)
    float*    sunrel = (float*)   (ws + 18 * MB);    // 32 MiB  (16384*512 f32)
    _Float16* srel   = (_Float16*)(ws + 50 * MB);    // 16 MiB  (16384*512 f16)
    int*      off    = (int*)     (ws + 66 * MB);    // 9 ints
    int*      idxl   = off + 16;                     // 16384 ints

    const int nDecW = 4 * D * D;
    const int nExpW = NEXP * 4 * D * D;
    st_cvt_f16<<<(nDecW + 255) / 256, 256, 0, stream>>>(decW, W16dec, nDecW);
    st_cvt_f16<<<(nExpW + 255) / 256, 256, 0, stream>>>(expW, W16exp, nExpW);
    st_partition<<<1, 1024, 0, stream>>>(y, off, idxl);
    st_decomposer<<<BATCH / BT, 256, 0, stream>>>(s, W16dec, decb, sunrel, srel);
    st_expert<<<NEXP * MAXTILES, 256, 0, stream>>>(srel, sunrel, W16exp, expb, off, idxl, out);
}
